// gw_GUR_cell_33681133535635
// MI455X (gfx1250) — matrix operand order repaired and checked by reading, not compiled, not run on hardware
//
#include <hip/hip_runtime.h>
#include <cstdint>
#include <cstddef>

// ---------------- problem constants ----------------
#define NNODE 512
#define BSZ   128
#define FIN   2
#define COUT  64
#define CIN   66                       // FIN + COUT
#define BFCOL (BSZ * CIN)              // 8448 columns of diffusion GEMM
#define SLOT  ((size_t)NNODE * BFCOL)  // 4,325,376 bf16 elems per slot
#define MROWS (NNODE * BSZ)            // 65536 rows of MLP GEMM
#define KPAD  480                      // 462 padded to multiple of 32
#define KREAL 462

typedef __attribute__((ext_vector_type(16))) __bf16 bf16x16;
typedef __attribute__((ext_vector_type(8)))  float  floatx8;

__device__ __forceinline__ __bf16 f2bf(float x) { return (__bf16)x; }

// LDS byte offset of a __shared__ pointer (flat LDS aperture: low 32 bits = offset)
__device__ __forceinline__ unsigned lds_off(const void* p) {
  return (unsigned)(size_t)p;
}

// ---------------- small prep kernels ----------------

// G[m][n] = bf16(A[n][m])   (A is [512,512] fp32, G is A^T in bf16)
__global__ void transpose_to_bf16_kernel(const float* __restrict__ A,
                                         __bf16* __restrict__ G) {
  int idx = blockIdx.x * 256 + threadIdx.x;     // < 512*512
  int m = idx >> 9;
  int n = idx & 511;
  G[idx] = f2bf(A[n * 512 + m]);
}

// adp = softmax(relu(nv1 @ nv2), axis=1); write G[m][n] = bf16(adp[n][m])
__global__ void adp_kernel(const float* __restrict__ nv1,
                           const float* __restrict__ nv2,
                           __bf16* __restrict__ G) {
  const int n = blockIdx.x;        // row of adp
  const int t = threadIdx.x;       // 256 threads; m = t and t+256
  __shared__ float red[256];

  float a[10];
#pragma unroll
  for (int k = 0; k < 10; ++k) a[k] = nv1[n * 10 + k];

  float v0 = 0.f, v1 = 0.f;
#pragma unroll
  for (int k = 0; k < 10; ++k) {
    v0 += a[k] * nv2[k * NNODE + t];
    v1 += a[k] * nv2[k * NNODE + t + 256];
  }
  v0 = fmaxf(v0, 0.f);
  v1 = fmaxf(v1, 0.f);

  red[t] = fmaxf(v0, v1);
  __syncthreads();
  for (int s = 128; s > 0; s >>= 1) {
    if (t < s) red[t] = fmaxf(red[t], red[t + s]);
    __syncthreads();
  }
  float mx = red[0];
  __syncthreads();

  float e0 = __expf(v0 - mx), e1 = __expf(v1 - mx);
  red[t] = e0 + e1;
  __syncthreads();
  for (int s = 128; s > 0; s >>= 1) {
    if (t < s) red[t] += red[t + s];
    __syncthreads();
  }
  float inv = 1.f / red[0];

  G[(size_t)t * NNODE + n]         = f2bf(e0 * inv);
  G[(size_t)(t + 256) * NNODE + n] = f2bf(e1 * inv);
}

// X[n][b*66+f] = bf16( f<2 ? input[b][n][f] : (use_r ? r : 1) * state[b][n][f-2] )
__global__ void pack_x_kernel(const float* __restrict__ input,
                              const float* __restrict__ state,
                              const float* __restrict__ value, int use_r,
                              __bf16* __restrict__ X) {
  int idx = blockIdx.x * 256 + threadIdx.x;     // < NNODE*BFCOL
  int n = idx / BFCOL;
  int j = idx - n * BFCOL;
  int b = j / CIN;
  int f = j - b * CIN;
  float v;
  if (f < FIN) {
    v = input[((size_t)b * NNODE + n) * FIN + f];
  } else {
    int c = f - FIN;
    v = state[((size_t)b * NNODE + n) * COUT + c];
    if (use_r) v *= value[((size_t)n * BSZ + b) * (2 * COUT) + c];
  }
  X[idx] = f2bf(v);
}

// H[row][kk] : row = n*128+b; kk = s*66+f -> slots[s][n][b*66+f]; pad K to 480
__global__ void pack_h_kernel(const __bf16* __restrict__ slots,
                              __bf16* __restrict__ H) {
  size_t idx = (size_t)blockIdx.x * 256 + threadIdx.x;   // < MROWS*KPAD
  int row = (int)(idx / KPAD);
  int kk  = (int)(idx - (size_t)row * KPAD);
  __bf16 v = f2bf(0.f);
  if (kk < KREAL) {
    int s = kk / CIN;
    int f = kk - s * CIN;
    int n = row >> 7;
    int b = row & 127;
    v = slots[(size_t)s * SLOT + (size_t)n * BFCOL + b * CIN + f];
  }
  H[idx] = v;
}

// Wp[k][j] = k<462 ? bf16(W[k][j]) : 0     (K padded to 480)
__global__ void pack_w_kernel(const float* __restrict__ W,
                              __bf16* __restrict__ Wp, int nc) {
  int idx = blockIdx.x * 256 + threadIdx.x;    // < KPAD*nc
  int k = idx / nc;
  int j = idx - k * nc;
  Wp[idx] = f2bf(k < KREAL ? W[k * nc + j] : 0.f);
}

// ---------------- the WMMA GEMM kernel ----------------
// C[M,Nc] = A[M,K] * B[K,Nc], bf16 row-major, f32 accumulate.
// Block: 128 threads (4 waves). Block tile 128x64, wave tile 32x64, K-step 32.
// Double-buffered LDS; A tile staged via GLOBAL_LOAD_ASYNC_TO_LDS_B128 (TDM-free
// async DMA path, ASYNCcnt), B tile register-staged + transposed into LDS so
// each WMMA fragment read is a contiguous ds_load_b128 pair.
// mode 0: store bf16 (diffusion);  mode 1: sigmoid -> f32 (gates)
// mode 2: tanh + GRU combine -> f32 output
__global__ __launch_bounds__(128)
void gemm_bf16_kernel(const __bf16* __restrict__ A, int lda,
                      const __bf16* __restrict__ Bm, int ldb, int K,
                      __bf16* __restrict__ Cbf, float* __restrict__ Cf, int ldc,
                      int mode,
                      const float* __restrict__ bias,
                      const float* __restrict__ value,
                      const float* __restrict__ state,
                      float* __restrict__ outp) {
  const int n0 = blockIdx.x * 64;
  const int m0 = blockIdx.y * 128;
  const int tid  = threadIdx.x;
  const int wave = tid >> 5;
  const int lane = tid & 31;
  const int half = lane >> 4;
  const int l16  = lane & 15;

  __shared__ __bf16 As[2][128 * 32];   // row-major [m][k], 8KB each
  __shared__ __bf16 Bs[2][64 * 32];    // transposed [n][k], 4KB each

  floatx8 acc[2][4];
#pragma unroll
  for (int i = 0; i < 2; ++i)
#pragma unroll
    for (int s = 0; s < 4; ++s)
#pragma unroll
      for (int j = 0; j < 8; ++j) acc[i][s][j] = 0.f;

  const int nk = K >> 5;
  uint4 breg[2];

  // --- async A tile: thread t copies row (m0+t), 4 x 16B, global -> LDS ---
  auto async_a = [&](int p, int k0) {
    const __bf16* gp = A + (size_t)(m0 + tid) * lda + k0;
    unsigned la = lds_off(&As[p][tid * 32]);
#pragma unroll
    for (int c = 0; c < 4; ++c) {
      unsigned long long ga = (unsigned long long)(size_t)(gp + c * 8);
      unsigned lc = la + c * 16;
      asm volatile("global_load_async_to_lds_b128 %0, %1, off"
                   :: "v"(lc), "v"(ga) : "memory");
    }
  };
  // --- B tile: 2 x 8 contiguous bf16 per thread into registers ---
  auto load_b = [&](int k0) {
#pragma unroll
    for (int i = 0; i < 2; ++i) {
      int c  = tid * 2 + i;
      int k  = c >> 3;
      int nb = (c & 7) << 3;
      breg[i] = *(const uint4*)(Bm + (size_t)(k0 + k) * ldb + n0 + nb);
    }
  };
  auto store_b = [&](int p) {
#pragma unroll
    for (int i = 0; i < 2; ++i) {
      int c  = tid * 2 + i;
      int k  = c >> 3;
      int nb = (c & 7) << 3;
      const __bf16* e = (const __bf16*)&breg[i];
#pragma unroll
      for (int j = 0; j < 8; ++j) Bs[p][(nb + j) * 32 + k] = e[j];
    }
  };
  auto compute = [&](int p) {
    bf16x16 bfv[4];
#pragma unroll
    for (int s = 0; s < 4; ++s) {
      ((uint4*)&bfv[s])[0] = *(const uint4*)&Bs[p][(s * 16 + l16) * 32 + half * 16];
      ((uint4*)&bfv[s])[1] = *(const uint4*)&Bs[p][(s * 16 + l16) * 32 + half * 16 + 8];
    }
#pragma unroll
    for (int i = 0; i < 2; ++i) {
      bf16x16 af;
      const int r = (wave * 32 + i * 16 + l16) * 32;
      ((uint4*)&af)[0] = *(const uint4*)&As[p][r + half * 16];
      ((uint4*)&af)[1] = *(const uint4*)&As[p][r + half * 16 + 8];
#pragma unroll
      for (int s = 0; s < 4; ++s)
        acc[i][s] = __builtin_amdgcn_wmma_f32_16x16x32_bf16(
            false, af, false, bfv[s], (short)0, acc[i][s], false, false);
    }
  };

  // prologue
  async_a(0, 0);
  load_b(0);
  store_b(0);
  asm volatile("s_wait_asynccnt 0x0" ::: "memory");
  __syncthreads();

  int p = 0;
  for (int kt = 0; kt < nk; ++kt) {
    const bool more = (kt + 1 < nk);
    if (more) {
      async_a(p ^ 1, (kt + 1) << 5);   // DMA next A tile (overlaps WMMA)
      load_b((kt + 1) << 5);           // next B tile in flight during compute
    }
    compute(p);
    if (more) {
      store_b(p ^ 1);
      asm volatile("s_wait_asynccnt 0x0" ::: "memory");
    }
    __syncthreads();
    p ^= 1;
  }

  // epilogue: component j of v8f -> row (half*8 + j), lane l16 -> column
#pragma unroll
  for (int i = 0; i < 2; ++i) {
    const int rbase = m0 + wave * 32 + i * 16 + half * 8;
#pragma unroll
    for (int s = 0; s < 4; ++s) {
      const int col = n0 + s * 16 + l16;
      if (mode == 0) {
#pragma unroll
        for (int j = 0; j < 8; ++j)
          Cbf[(size_t)(rbase + j) * ldc + col] = f2bf(acc[i][s][j]);
      } else if (mode == 1) {
        float bb = bias[col];
#pragma unroll
        for (int j = 0; j < 8; ++j) {
          float v = acc[i][s][j] + bb;
          Cf[(size_t)(rbase + j) * ldc + col] = 1.f / (1.f + __expf(-v));
        }
      } else {
        float bb = bias[col];
#pragma unroll
        for (int j = 0; j < 8; ++j) {
          int row = rbase + j;                 // row = n*128 + b
          float cval = tanhf(acc[i][s][j] + bb);
          int n = row >> 7;
          int b = row & 127;
          size_t oi = ((size_t)b * NNODE + n) * COUT + col;
          float u = value[(size_t)row * (2 * COUT) + COUT + col];
          outp[oi] = u * state[oi] + (1.f - u) * cval;
        }
      }
    }
  }
}

// ---------------- host-side orchestration ----------------

static inline void launch_diffusion(const __bf16* G, const __bf16* Bx,
                                    __bf16* Y, hipStream_t stream) {
  dim3 grid(BFCOL / 64, NNODE / 128);   // 132 x 4
  gemm_bf16_kernel<<<grid, 128, 0, stream>>>(G, NNODE, Bx, BFCOL, NNODE,
                                             Y, nullptr, BFCOL, 0,
                                             nullptr, nullptr, nullptr, nullptr);
}

extern "C" void kernel_launch(void* const* d_in, const int* in_sizes, int n_in,
                              void* d_out, int out_size, void* d_ws, size_t ws_size,
                              hipStream_t stream) {
  const float* input = (const float*)d_in[0];
  const float* state = (const float*)d_in[1];
  const float* A0    = (const float*)d_in[2];
  const float* A1    = (const float*)d_in[3];
  const float* nv1_1 = (const float*)d_in[4];
  const float* nv2_1 = (const float*)d_in[5];
  const float* W1    = (const float*)d_in[6];
  const float* b1    = (const float*)d_in[7];
  const float* nv1_2 = (const float*)d_in[8];
  const float* nv2_2 = (const float*)d_in[9];
  const float* W2    = (const float*)d_in[10];
  const float* b2    = (const float*)d_in[11];
  float* outp = (float*)d_out;

  char* ws = (char*)d_ws;
  // workspace layout (bytes)
  const size_t GBYTES = (size_t)NNODE * NNODE * 2;          // 512 KB
  __bf16* GA0   = (__bf16*)(ws + 0 * GBYTES);
  __bf16* GA1   = (__bf16*)(ws + 1 * GBYTES);
  __bf16* Gadp1 = (__bf16*)(ws + 2 * GBYTES);
  __bf16* Gadp2 = (__bf16*)(ws + 3 * GBYTES);
  size_t off = 4 * GBYTES;
  __bf16* slots = (__bf16*)(ws + off);                      // 7 slots
  off += 7 * SLOT * 2;
  __bf16* Hpad = (__bf16*)(ws + off);                       // [65536,480]
  off += (size_t)MROWS * KPAD * 2;
  __bf16* Wpad = (__bf16*)(ws + off);                       // [480,128] max
  off += (size_t)KPAD * 128 * 2;
  float* value = (float*)(ws + off);                        // [65536,128]

  __bf16* X0 = slots;  // slot 0

  // adjacency operands (bf16, transposed)
  transpose_to_bf16_kernel<<<(NNODE * NNODE) / 256, 256, 0, stream>>>(A0, GA0);
  transpose_to_bf16_kernel<<<(NNODE * NNODE) / 256, 256, 0, stream>>>(A1, GA1);
  adp_kernel<<<NNODE, 256, 0, stream>>>(nv1_1, nv2_1, Gadp1);
  adp_kernel<<<NNODE, 256, 0, stream>>>(nv1_2, nv2_2, Gadp2);

  const int PACKX_BLOCKS = (int)(SLOT / 256);       // 16896
  const int PACKH_BLOCKS = (MROWS * KPAD) / 256;    // 122880

  // ---- GCN 1 ----
  pack_x_kernel<<<PACKX_BLOCKS, 256, 0, stream>>>(input, state, nullptr, 0, X0);
  launch_diffusion(GA0,   X0,               slots + 1 * SLOT, stream);
  launch_diffusion(GA0,   slots + 1 * SLOT, slots + 2 * SLOT, stream);
  launch_diffusion(GA1,   X0,               slots + 3 * SLOT, stream);
  launch_diffusion(GA1,   slots + 3 * SLOT, slots + 4 * SLOT, stream);
  launch_diffusion(Gadp1, X0,               slots + 5 * SLOT, stream);
  launch_diffusion(Gadp1, slots + 5 * SLOT, slots + 6 * SLOT, stream);

  pack_h_kernel<<<PACKH_BLOCKS, 256, 0, stream>>>(slots, Hpad);
  pack_w_kernel<<<(KPAD * 128) / 256, 256, 0, stream>>>(W1, Wpad, 128);
  {
    dim3 grid(128 / 64, MROWS / 128);   // 2 x 512
    gemm_bf16_kernel<<<grid, 128, 0, stream>>>(Hpad, KPAD, Wpad, 128, KPAD,
                                               nullptr, value, 128, 1,
                                               b1, nullptr, nullptr, nullptr);
  }

  // ---- GCN 2 (input gated by r) ----
  pack_x_kernel<<<PACKX_BLOCKS, 256, 0, stream>>>(input, state, value, 1, X0);
  launch_diffusion(GA0,   X0,               slots + 1 * SLOT, stream);
  launch_diffusion(GA0,   slots + 1 * SLOT, slots + 2 * SLOT, stream);
  launch_diffusion(GA1,   X0,               slots + 3 * SLOT, stream);
  launch_diffusion(GA1,   slots + 3 * SLOT, slots + 4 * SLOT, stream);
  launch_diffusion(Gadp2, X0,               slots + 5 * SLOT, stream);
  launch_diffusion(Gadp2, slots + 5 * SLOT, slots + 6 * SLOT, stream);

  pack_h_kernel<<<PACKH_BLOCKS, 256, 0, stream>>>(slots, Hpad);
  pack_w_kernel<<<(KPAD * COUT) / 256, 256, 0, stream>>>(W2, Wpad, COUT);
  {
    dim3 grid(COUT / 64, MROWS / 128);  // 1 x 512
    gemm_bf16_kernel<<<grid, 128, 0, stream>>>(Hpad, KPAD, Wpad, COUT, KPAD,
                                               nullptr, nullptr, COUT, 2,
                                               b2, value, state, outp);
  }
}
